// MoE_13116830122699
// MI455X (gfx1250) — compile-verified, matrix operand order checked by reading, not run on hardware
//
#include <hip/hip_runtime.h>

#define T_TOK  4096
#define DMODEL 2048
#define NEXP   8
#define FFN    4096   // F == FS in this problem

typedef __attribute__((ext_vector_type(16))) __bf16 bf16x16;
typedef __attribute__((ext_vector_type(8)))  float  v8f;
typedef __attribute__((ext_vector_type(4)))  unsigned int u32x4;
typedef __attribute__((ext_vector_type(2)))  unsigned int u32x2;

union FragU { bf16x16 f; u32x4 u[2]; };

// fp32 -> bf16, round-to-nearest-even
__device__ __forceinline__ unsigned int f2bf(float x) {
  unsigned int u = __float_as_uint(x);
  u += 0x7fffu + ((u >> 16) & 1u);
  return u >> 16;
}

// Async 16B global -> LDS copy (CDNA5 GLOBAL_LOAD_ASYNC_TO_LDS_B128, ASYNCcnt).
// LDS dest address = low 32 bits of the flat shared-aperture pointer
// (ISA 10.2: LDS aperture keeps the LDS offset in addr[31:0]).
__device__ __forceinline__ void async_cp16(unsigned short* ldsDst,
                                           const unsigned short* gSrc) {
  unsigned loff = (unsigned)(unsigned long long)(uintptr_t)ldsDst;
  unsigned long long ga = (unsigned long long)(uintptr_t)gSrc;
  asm volatile("global_load_async_to_lds_b128 %0, %1, off"
               :: "v"(loff), "v"(ga) : "memory");
}

__device__ __forceinline__ void wait_async0() {
  asm volatile("s_wait_asynccnt 0x0" ::: "memory");
}

// LDS tile rows are 32 bf16 of K padded to 40 shorts (80B) for bank-conflict-free b128 loads.
// Fragment layout per CDNA5 ISA (16-bit A/B 16x32): lanes 0-15 hold K 0-7 & 16-23,
// lanes 16-31 hold K 8-15 & 24-31 of their row.
__device__ __forceinline__ bf16x16 ldsFrag(const unsigned short* base, int row, int hl) {
  FragU fr;
  fr.u[0] = *(const u32x4*)(base + row * 40 + hl * 8);
  fr.u[1] = *(const u32x4*)(base + row * 40 + 16 + hl * 8);
  return fr.f;
}

// ---------------------------------------------------------------------------
// x (fp32) -> bf16 workspace copy
// ---------------------------------------------------------------------------
__global__ __launch_bounds__(256) void convert_bf16(const float* __restrict__ in,
                                                    unsigned short* __restrict__ outp) {
  size_t i = (size_t)blockIdx.x * 256 + threadIdx.x;   // one float4 per thread, exact tiling
  float4 v = ((const float4*)in)[i];
  u32x2 p;
  p[0] = f2bf(v.x) | (f2bf(v.y) << 16);
  p[1] = f2bf(v.z) | (f2bf(v.w) << 16);
  ((u32x2*)outp)[i] = p;
}

// ---------------------------------------------------------------------------
// Router: logits = x @ router_DE, top-2, renormalized softmax -> dense cw[T, E]
// One wave32 per token; butterfly reduction; lanes 0..7 write their expert slot.
// ---------------------------------------------------------------------------
__global__ __launch_bounds__(256) void router_topk(const float* __restrict__ x,
                                                   const float* __restrict__ rdE,
                                                   float* __restrict__ cw) {
  const int lane = threadIdx.x & 31;
  const int wv   = threadIdx.x >> 5;
  const int t    = blockIdx.x * 8 + wv;

  float lg[8];
#pragma unroll
  for (int e = 0; e < 8; ++e) lg[e] = 0.0f;

  for (int d = lane; d < DMODEL; d += 32) {
    float xv = x[(size_t)t * DMODEL + d];
    const float* rr = rdE + (size_t)d * NEXP;
    float4 r0 = *(const float4*)(rr);
    float4 r1 = *(const float4*)(rr + 4);
    lg[0] += xv * r0.x; lg[1] += xv * r0.y; lg[2] += xv * r0.z; lg[3] += xv * r0.w;
    lg[4] += xv * r1.x; lg[5] += xv * r1.y; lg[6] += xv * r1.z; lg[7] += xv * r1.w;
  }

#pragma unroll
  for (int off = 16; off > 0; off >>= 1)
#pragma unroll
    for (int e = 0; e < 8; ++e)
      lg[e] += __shfl_xor(lg[e], off, 32);

  // top-2 (ties -> lower index, matching lax.top_k)
  int i0 = 0; float v0 = lg[0];
#pragma unroll
  for (int e = 1; e < 8; ++e) if (lg[e] > v0) { v0 = lg[e]; i0 = e; }
  int i1 = -1; float v1 = -3.402823466e38f;
#pragma unroll
  for (int e = 0; e < 8; ++e) if (e != i0 && lg[e] > v1) { v1 = lg[e]; i1 = e; }

  float e1 = __expf(v1 - v0);
  float g0 = 1.0f / (1.0f + e1);
  float g1 = e1 * g0;

  if (lane < 8) {
    float val = (lane == i0) ? g0 : ((lane == i1) ? g1 : 0.0f);
    cw[(size_t)t * NEXP + lane] = val;
  }
}

// ---------------------------------------------------------------------------
// Fused gate/up GEMM + SwiGLU, double-buffered LDS, async A-tile copies:
//   C1 = x @ w13[0:F,:].T, C2 = x @ w13[F:2F,:].T, a = silu(C1)*C2 -> bf16
// Block tile 128(M tokens) x 128(N features), K step 32, 8 waves (2x4),
// wave tile 64x32 -> 4x2 accum pairs, 16 v_wmma_f32_16x16x32_bf16 per K step.
// ---------------------------------------------------------------------------
__global__ __launch_bounds__(256) void gemm_gate_up(const unsigned short* __restrict__ xbf,
                                                    const float* __restrict__ w13,
                                                    unsigned short* __restrict__ aout) {
  __shared__ __align__(16) unsigned short lA [2][128 * 40];
  __shared__ __align__(16) unsigned short lB1[2][128 * 40];
  __shared__ __align__(16) unsigned short lB2[2][128 * 40];

  const int tid   = threadIdx.x;
  const int lane  = tid & 31, wv = tid >> 5;
  const int waveM = wv >> 2,  waveN = wv & 3;
  const int hl    = lane >> 4, lr = lane & 15;
  const int t0    = blockIdx.y * 128;
  const int n0    = blockIdx.x * 128;

  // A tile: 128x32 bf16, async DMA into LDS (no VGPR round-trip)
  auto fillA = [&](int kt, int buf) {
#pragma unroll
    for (int i = 0; i < 2; ++i) {
      int c = tid + i * 256;            // 512 chunks of 8 shorts
      int row = c >> 2, cc = c & 3;
      async_cp16(&lA[buf][row * 40 + cc * 8],
                 xbf + (size_t)(t0 + row) * DMODEL + kt + cc * 8);
    }
  };
  // B tiles: 128x32 fp32 -> bf16 on the fly (gate half and up half of w13)
  auto fillB = [&](int kt, int buf) {
#pragma unroll
    for (int i = 0; i < 4; ++i) {
      int c = tid + i * 256;            // 1024 groups of 4 floats
      int row = c >> 3, cc = c & 7;
      const float* g1 = w13 + (size_t)(n0 + row) * DMODEL + kt + cc * 4;
      const float* g2 = w13 + (size_t)(FFN + n0 + row) * DMODEL + kt + cc * 4;
      if (i == 0 && kt + 32 < DMODEL) {
        __builtin_prefetch(g1 + 32, 0, 3);
        __builtin_prefetch(g2 + 32, 0, 3);
      }
      float4 v1 = *(const float4*)g1;
      float4 v2 = *(const float4*)g2;
      u32x2 p1, p2;
      p1[0] = f2bf(v1.x) | (f2bf(v1.y) << 16);
      p1[1] = f2bf(v1.z) | (f2bf(v1.w) << 16);
      p2[0] = f2bf(v2.x) | (f2bf(v2.y) << 16);
      p2[1] = f2bf(v2.z) | (f2bf(v2.w) << 16);
      *(u32x2*)(&lB1[buf][row * 40 + cc * 4]) = p1;
      *(u32x2*)(&lB2[buf][row * 40 + cc * 4]) = p2;
    }
  };

  v8f acc1[4][2] = {};
  v8f acc2[4][2] = {};

  fillA(0, 0);
  fillB(0, 0);
  wait_async0();
  __syncthreads();

  const int NK = DMODEL / 32;
  for (int k = 0; k < NK; ++k) {
    const int cur = k & 1;
    // Prefetch next tiles into the other buffer; its readers finished at the
    // barrier that ended iteration k-1, so this overlaps with the WMMAs below.
    if (k + 1 < NK) {
      fillA((k + 1) * 32, cur ^ 1);
      fillB((k + 1) * 32, cur ^ 1);
    }

    bf16x16 af[4], bf[2];
#pragma unroll
    for (int tm = 0; tm < 4; ++tm)
      af[tm] = ldsFrag(lA[cur], waveM * 64 + tm * 16 + lr, hl);

#pragma unroll
    for (int tn = 0; tn < 2; ++tn)
      bf[tn] = ldsFrag(lB1[cur], waveN * 32 + tn * 16 + lr, hl);
#pragma unroll
    for (int tm = 0; tm < 4; ++tm)
#pragma unroll
      for (int tn = 0; tn < 2; ++tn)
        acc1[tm][tn] = __builtin_amdgcn_wmma_f32_16x16x32_bf16(
            false, af[tm], false, bf[tn], (short)0, acc1[tm][tn], false, false);

#pragma unroll
    for (int tn = 0; tn < 2; ++tn)
      bf[tn] = ldsFrag(lB2[cur], waveN * 32 + tn * 16 + lr, hl);
#pragma unroll
    for (int tm = 0; tm < 4; ++tm)
#pragma unroll
      for (int tn = 0; tn < 2; ++tn)
        acc2[tm][tn] = __builtin_amdgcn_wmma_f32_16x16x32_bf16(
            false, af[tm], false, bf[tn], (short)0, acc2[tm][tn], false, false);

    wait_async0();        // async A-tile writes for buffer cur^1 are in LDS
    __syncthreads();      // publish next buffer, retire reads of this one
  }

  // SwiGLU epilogue -> bf16 activations
#pragma unroll
  for (int tm = 0; tm < 4; ++tm)
#pragma unroll
    for (int tn = 0; tn < 2; ++tn) {
      int col = n0 + waveN * 32 + tn * 16 + lr;
#pragma unroll
      for (int i = 0; i < 8; ++i) {
        int row = t0 + waveM * 64 + tm * 16 + hl * 8 + i;  // C layout: VGPR i -> M=i(+8)
        float c1 = acc1[tm][tn][i];
        float c2 = acc2[tm][tn][i];
        float sw = c1 / (1.0f + __expf(-c1));              // silu
        aout[(size_t)row * FFN + col] = (unsigned short)f2bf(sw * c2);
      }
    }
}

// ---------------------------------------------------------------------------
// Down projection: out[t,:] (+)= gate[t] * (a @ w2.T)
// gate == nullptr -> 1.0 (shared expert); accumulate==0 initializes out.
// ---------------------------------------------------------------------------
__global__ __launch_bounds__(256) void gemm_down(const unsigned short* __restrict__ abf,
                                                 const float* __restrict__ w2,
                                                 float* __restrict__ out,
                                                 const float* __restrict__ gate,
                                                 int accumulate) {
  __shared__ __align__(16) unsigned short lA[2][128 * 40];
  __shared__ __align__(16) unsigned short lB[2][128 * 40];

  const int tid   = threadIdx.x;
  const int lane  = tid & 31, wv = tid >> 5;
  const int waveM = wv >> 2,  waveN = wv & 3;
  const int hl    = lane >> 4, lr = lane & 15;
  const int t0    = blockIdx.y * 128;
  const int n0    = blockIdx.x * 128;   // over DMODEL

  auto fillA = [&](int kt, int buf) {
#pragma unroll
    for (int i = 0; i < 2; ++i) {
      int c = tid + i * 256;
      int row = c >> 2, cc = c & 3;
      async_cp16(&lA[buf][row * 40 + cc * 8],
                 abf + (size_t)(t0 + row) * FFN + kt + cc * 8);
    }
  };
  auto fillB = [&](int kt, int buf) {
#pragma unroll
    for (int i = 0; i < 4; ++i) {
      int c = tid + i * 256;
      int row = c >> 3, cc = c & 7;
      const float* gp = w2 + (size_t)(n0 + row) * FFN + kt + cc * 4;
      if (i == 0 && kt + 32 < FFN) __builtin_prefetch(gp + 32, 0, 3);
      float4 v = *(const float4*)gp;
      u32x2 p;
      p[0] = f2bf(v.x) | (f2bf(v.y) << 16);
      p[1] = f2bf(v.z) | (f2bf(v.w) << 16);
      *(u32x2*)(&lB[buf][row * 40 + cc * 4]) = p;
    }
  };

  v8f acc[4][2] = {};

  fillA(0, 0);
  fillB(0, 0);
  wait_async0();
  __syncthreads();

  const int NK = FFN / 32;
  for (int k = 0; k < NK; ++k) {
    const int cur = k & 1;
    if (k + 1 < NK) {
      fillA((k + 1) * 32, cur ^ 1);
      fillB((k + 1) * 32, cur ^ 1);
    }

    bf16x16 af[4], bf[2];
#pragma unroll
    for (int tm = 0; tm < 4; ++tm)
      af[tm] = ldsFrag(lA[cur], waveM * 64 + tm * 16 + lr, hl);
#pragma unroll
    for (int tn = 0; tn < 2; ++tn)
      bf[tn] = ldsFrag(lB[cur], waveN * 32 + tn * 16 + lr, hl);
#pragma unroll
    for (int tm = 0; tm < 4; ++tm)
#pragma unroll
      for (int tn = 0; tn < 2; ++tn)
        acc[tm][tn] = __builtin_amdgcn_wmma_f32_16x16x32_bf16(
            false, af[tm], false, bf[tn], (short)0, acc[tm][tn], false, false);

    wait_async0();
    __syncthreads();
  }

#pragma unroll
  for (int tm = 0; tm < 4; ++tm)
#pragma unroll
    for (int tn = 0; tn < 2; ++tn) {
      int col = n0 + waveN * 32 + tn * 16 + lr;
#pragma unroll
      for (int i = 0; i < 8; ++i) {
        int row = t0 + waveM * 64 + tm * 16 + hl * 8 + i;
        float g = gate ? gate[(size_t)row * NEXP] : 1.0f;   // gate pre-offset by expert id
        float v = g * acc[tm][tn][i];
        size_t o = (size_t)row * DMODEL + col;
        out[o] = accumulate ? (out[o] + v) : v;
      }
    }
}

// ---------------------------------------------------------------------------
extern "C" void kernel_launch(void* const* d_in, const int* in_sizes, int n_in,
                              void* d_out, int out_size, void* d_ws, size_t ws_size,
                              hipStream_t stream) {
  (void)in_sizes; (void)n_in; (void)out_size; (void)ws_size;

  const float* x    = (const float*)d_in[0];
  const float* rdE  = (const float*)d_in[1];
  const float* w13  = (const float*)d_in[2];   // [E, 2F, D]
  const float* w2   = (const float*)d_in[3];   // [E, D, F]
  const float* sw13 = (const float*)d_in[4];   // [2F, D]
  const float* sw2  = (const float*)d_in[5];   // [D, F]
  float* out = (float*)d_out;

  char* ws = (char*)d_ws;
  float*          cw  = (float*)ws;                                   // T*E fp32 (128 KB)
  unsigned short* xbf = (unsigned short*)(ws + 131072);               // T*D bf16 (16.8 MB)
  unsigned short* abf = (unsigned short*)(ws + 131072 + (size_t)T_TOK * DMODEL * 2); // T*F bf16

  convert_bf16<<<(T_TOK * DMODEL) / 1024, 256, 0, stream>>>(x, xbf);
  router_topk<<<T_TOK / 8, 256, 0, stream>>>(x, rdE, cw);

  dim3 block(256);
  dim3 gridGU(FFN / 128, T_TOK / 128);     // 32 x 32
  dim3 gridDN(DMODEL / 128, T_TOK / 128);  // 16 x 32

  // shared expert first: initializes out (accumulate=0, gate=1.0)
  gemm_gate_up<<<gridGU, block, 0, stream>>>(xbf, sw13, abf);
  gemm_down  <<<gridDN, block, 0, stream>>>(abf, sw2, out, nullptr, 0);

  // routed experts in fixed order: deterministic accumulation
  for (int e = 0; e < NEXP; ++e) {
    const float* w13e = w13 + (size_t)e * 2 * FFN * DMODEL;
    const float* w2e  = w2  + (size_t)e * DMODEL * FFN;
    gemm_gate_up<<<gridGU, block, 0, stream>>>(xbf, w13e, abf);
    gemm_down  <<<gridDN, block, 0, stream>>>(abf, w2e, out, cw + e, 1);
  }
}
